// Generator_2748779069790
// MI455X (gfx1250) — matrix operand order repaired and checked by reading, not compiled, not run on hardware
//
#include <hip/hip_runtime.h>

#define BATCH   16384
#define HID     256
#define SEQ     20
#define OUTC    214
#define OUTPAD  224
#define TB      128          // batch rows per workgroup
#define H_STRIDE 264         // padded LDS row stride (bf16 elements), 528B = 33*16B
#define BUF_ELEMS (TB * H_STRIDE)
#define CSTRIDE  260         // padded LDS row stride for fp32 cell state

typedef __bf16 bf16_t;
typedef bf16_t v16bf __attribute__((ext_vector_type(16)));
typedef float  v8f   __attribute__((ext_vector_type(8)));
typedef unsigned int v4u __attribute__((ext_vector_type(4)));

union Frag16 { v4u u[2]; v16bf v; };

// ---- helpers -------------------------------------------------------------

__device__ __forceinline__ unsigned short f2bf(float x) {
  unsigned u = __builtin_bit_cast(unsigned, x);
  unsigned r = u + 0x7FFFu + ((u >> 16) & 1u);   // round-to-nearest-even
  return (unsigned short)(r >> 16);
}

__device__ __forceinline__ float fsig(float x)  { return 1.0f / (1.0f + __expf(-x)); }
__device__ __forceinline__ float ftanh_(float x){ return 2.0f / (1.0f + __expf(-2.0f * x)) - 1.0f; }

// A fragment (16-bit A, 16x32): lane L<16 -> row L, K octets {0..7,16..23};
// lane L>=16 -> row L-16, K octets {8..15,24..31}.  Source: LDS h buffer.
__device__ __forceinline__ v16bf load_a_lds(const unsigned short* h, int rbase, int k0, int lane) {
  const int mr = lane & 15, half = lane >> 4;
  const unsigned short* p = h + (rbase + mr) * H_STRIDE + k0 + half * 8;
  Frag16 f;
  f.u[0] = *(const v4u*)(p);
  f.u[1] = *(const v4u*)(p + 16);
  return f.v;
}

// B fragment (16-bit B, 32x16): lane holds column (lane&15) = weight row,
// K octets {0..7,16..23} (lane<16) or {8..15,24..31} (lane>=16) along the weight row (row-major [*,256]).
__device__ __forceinline__ v16bf load_b_row(const unsigned short* W, int nrow, int k0, int lane) {
  const int n = lane & 15, kh = lane >> 4;
  const unsigned short* p = W + (size_t)(nrow + n) * HID + k0 + kh * 8;
  Frag16 f;
  f.u[0] = *(const v4u*)(p);
  f.u[1] = *(const v4u*)(p + 16);
  return f.v;
}

__device__ __forceinline__ v8f wmma_bf16(v16bf a, v16bf b, v8f c) {
  return __builtin_amdgcn_wmma_f32_16x16x32_bf16(false, a, false, b, (short)0, c, false, false);
}

// ---- prep: bf16 weights + fused bias vectors -----------------------------

__global__ void prep_kernel(
    const float* __restrict__ W_ih, const float* __restrict__ W_hh,
    const float* __restrict__ b_ih, const float* __restrict__ b_hh,
    const float* __restrict__ W_tech, const float* __restrict__ b_tech,
    const float* __restrict__ W_phase, const float* __restrict__ b_phase,
    const float* __restrict__ W_time, const float* __restrict__ b_time,
    unsigned short* __restrict__ wih, unsigned short* __restrict__ wcomb,
    unsigned short* __restrict__ wout, float* __restrict__ bg, float* __restrict__ bo) {
  int j = blockIdx.x * blockDim.x + threadIdx.x;
  if (j < 1024 * 256) {
    float a = W_ih[j], b = W_hh[j];
    wih[j]   = f2bf(a);
    wcomb[j] = f2bf(a + b);
  }
  if (j < OUTPAD * 256) {
    int row = j >> 8, col = j & 255;
    float v = 0.0f;
    if (row < 200)        v = W_tech[row * 256 + col];
    else if (row < 213)   v = W_phase[(row - 200) * 256 + col];
    else if (row == 213)  v = W_time[col];
    wout[j] = f2bf(v);
  }
  if (j < 1024) bg[j] = b_ih[j] + b_hh[j];
  if (j < OUTPAD) {
    float v = 0.0f;
    if (j < 200)       v = b_tech[j];
    else if (j < 213)  v = b_phase[j - 200];
    else if (j == 213) v = b_time[0];
    bo[j] = v;
  }
}

// ---- x0 = noise @ W_proj.T + b_proj  -> bf16 -----------------------------

__global__ __launch_bounds__(256) void proj_kernel(
    const float* __restrict__ noise, const float* __restrict__ Wp,
    const float* __restrict__ bp, unsigned short* __restrict__ x0) {
  __shared__ float nz[64][104];
  const int r0 = blockIdx.x * 64;
  for (int idx = threadIdx.x; idx < 64 * 100; idx += 256) {
    int r = idx / 100, k = idx - r * 100;
    nz[r][k] = noise[(size_t)(r0 + r) * 100 + k];
  }
  __syncthreads();
  const int c = threadIdx.x;   // output column 0..255
  float acc[64];
  #pragma unroll
  for (int r = 0; r < 64; ++r) acc[r] = 0.0f;
  for (int k = 0; k < 100; ++k) {
    float w = Wp[c * 100 + k];
    #pragma unroll
    for (int r = 0; r < 64; ++r) acc[r] += w * nz[r][k];
  }
  const float bb = bp[c];
  for (int r = 0; r < 64; ++r)
    x0[(size_t)(r0 + r) * HID + c] = f2bf(acc[r] + bb);
}

// ---- main persistent LSTM kernel -----------------------------------------
// LDS: h ping-pong (2 x 128 x 264 bf16 = 132 KB) + cell state (128 x 260 f32
// = 130 KB) = 262 KB of the WGP's 320 KB.  Cell state per wave touches only
// its own 32 columns -> no barrier needed for c.  K-loops unrolled by 2 to
// bound prefetch depth; A-fragments are loaded once per k and reused by all
// four gates while B-fragments stream one gate at a time, keeping the live
// set (~230 VGPRs) under the 256-VGPR MSB boundary.

__global__ __launch_bounds__(256)
__attribute__((amdgpu_waves_per_eu(2)))
void lstm_gen_kernel(
    const unsigned short* __restrict__ x0,
    const unsigned short* __restrict__ Wih,
    const unsigned short* __restrict__ Wcomb,
    const unsigned short* __restrict__ Wout,
    const float* __restrict__ bgate,
    const float* __restrict__ bout,
    float* __restrict__ out) {
  extern __shared__ unsigned short hbuf[];                  // 2 * BUF_ELEMS bf16
  float* cls = (float*)(hbuf + 2 * BUF_ELEMS);              // TB x CSTRIDE f32
  const int lane = threadIdx.x & 31;
  const int wave = threadIdx.x >> 5;
  const int n16  = lane & 15;
  const int half = lane >> 4;
  const int R0   = blockIdx.x * TB;
  const int cq0  = wave * 32;                               // this wave's 32 gate columns

  // stage x0 tile into buffer 0
  for (int idx = threadIdx.x; idx < TB * (HID / 8); idx += 256) {
    int row = idx >> 5;
    int c8  = (idx & 31) * 8;
    *(v4u*)&hbuf[row * H_STRIDE + c8] = *(const v4u*)&x0[(size_t)(R0 + row) * HID + c8];
  }
  // zero cell state
  for (int idx = threadIdx.x; idx < TB * CSTRIDE; idx += 256) cls[idx] = 0.0f;
  __syncthreads();

  float bgq[2][4];
  #pragma unroll
  for (int q = 0; q < 2; ++q) {
    int col = cq0 + q * 16 + n16;
    bgq[q][0] = bgate[col];
    bgq[q][1] = bgate[256 + col];
    bgq[q][2] = bgate[512 + col];
    bgq[q][3] = bgate[768 + col];
  }

  int cur = 0;
  #pragma unroll 1
  for (int t = 0; t < SEQ; ++t) {
    const unsigned short* Wg = (t == 0) ? Wih : Wcomb;
    const unsigned short* hc = hbuf + cur * BUF_ELEMS;
    unsigned short*       hn = hbuf + (cur ^ 1) * BUF_ELEMS;

    // gates + cell update + h_new for this wave's 32 columns, all 128 rows
    #pragma unroll
    for (int q = 0; q < 2; ++q) {
      const int cq = cq0 + q * 16;
      float* cbase = cls + (8 * half) * CSTRIDE + cq + n16;
      #pragma unroll
      for (int mg = 0; mg < 2; ++mg) {            // 2 groups of 4 M-tiles
        v8f acc[4][4];                            // [gate][mtile-in-group]
        #pragma unroll
        for (int g = 0; g < 4; ++g)
          #pragma unroll
          for (int ms = 0; ms < 4; ++ms)
            #pragma unroll
            for (int j = 0; j < 8; ++j) acc[g][ms][j] = bgq[q][g];
        #pragma unroll 2
        for (int k = 0; k < 8; ++k) {
          v16bf afr[4];                           // A resident, B streamed per gate
          #pragma unroll
          for (int ms = 0; ms < 4; ++ms)
            afr[ms] = load_a_lds(hc, (mg * 4 + ms) * 16, k * 32, lane);
          #pragma unroll
          for (int g = 0; g < 4; ++g) {
            v16bf b = load_b_row(Wg, g * 256 + cq, k * 32, lane);
            #pragma unroll
            for (int ms = 0; ms < 4; ++ms) acc[g][ms] = wmma_bf16(afr[ms], b, acc[g][ms]);
          }
        }
        #pragma unroll
        for (int ms = 0; ms < 4; ++ms) {
          const int m = mg * 4 + ms;
          #pragma unroll
          for (int j = 0; j < 8; ++j) {
            float iv = fsig(acc[0][ms][j]);
            float fv = fsig(acc[1][ms][j]);
            float gv = ftanh_(acc[2][ms][j]);
            float ov = fsig(acc[3][ms][j]);
            float* cp = cbase + (m * 16 + j) * CSTRIDE;
            float cv = fv * (*cp) + iv * gv;
            *cp = cv;
            float hv = ov * ftanh_(cv);
            hn[(m * 16 + j + 8 * half) * H_STRIDE + cq + n16] = f2bf(hv);
          }
        }
      }
    }
    __syncthreads();   // hn now holds h_{t+1} for all 128 rows

    // output head: out_t = h_new @ Wout.T + bout ; 14 column tiles over 8 waves
    float* outp = out + (size_t)t * BATCH * OUTC;
    for (int ot = wave; ot < 14; ot += 8) {
      const int nb  = ot * 16;
      const int col = nb + n16;
      const float bv = bout[col];
      #pragma unroll
      for (int mg = 0; mg < 2; ++mg) {
        v8f acc[4];
        #pragma unroll
        for (int ms = 0; ms < 4; ++ms)
          #pragma unroll
          for (int j = 0; j < 8; ++j) acc[ms][j] = bv;
        #pragma unroll 2
        for (int k = 0; k < 8; ++k) {
          v16bf b = load_b_row(Wout, nb, k * 32, lane);
          #pragma unroll
          for (int ms = 0; ms < 4; ++ms) {
            v16bf a = load_a_lds(hn, (mg * 4 + ms) * 16, k * 32, lane);
            acc[ms] = wmma_bf16(a, b, acc[ms]);
          }
        }
        if (col < OUTC) {
          #pragma unroll
          for (int ms = 0; ms < 4; ++ms) {
            #pragma unroll
            for (int j = 0; j < 8; ++j) {
              float v = acc[ms][j];
              if (col == 213) v = v > 0.0f ? v : 0.0f;   // relu(timing)
              int row = R0 + (mg * 4 + ms) * 16 + j + 8 * half;
              outp[(size_t)row * OUTC + col] = v;
            }
          }
        }
      }
    }
    cur ^= 1;
  }
}

// ---- host-side launch -----------------------------------------------------

extern "C" void kernel_launch(void* const* d_in, const int* in_sizes, int n_in,
                              void* d_out, int out_size, void* d_ws, size_t ws_size,
                              hipStream_t stream) {
  const float* noise   = (const float*)d_in[0];
  const float* W_proj  = (const float*)d_in[1];
  const float* b_proj  = (const float*)d_in[2];
  const float* W_ih    = (const float*)d_in[3];
  const float* W_hh    = (const float*)d_in[4];
  const float* b_ih    = (const float*)d_in[5];
  const float* b_hh    = (const float*)d_in[6];
  const float* W_tech  = (const float*)d_in[7];
  const float* b_tech  = (const float*)d_in[8];
  const float* W_phase = (const float*)d_in[9];
  const float* b_phase = (const float*)d_in[10];
  const float* W_time  = (const float*)d_in[11];
  const float* b_time  = (const float*)d_in[12];

  char* ws = (char*)d_ws;
  unsigned short* x0    = (unsigned short*)(ws);                                   // 8 MB
  unsigned short* wih   = (unsigned short*)(ws + 8388608);                         // 512 KB
  unsigned short* wcomb = (unsigned short*)(ws + 8388608 + 524288);                // 512 KB
  unsigned short* wout  = (unsigned short*)(ws + 8388608 + 2 * 524288);            // 112 KB
  float* bg             = (float*)(ws + 8388608 + 2 * 524288 + 114688);            // 4 KB
  float* bo             = (float*)(ws + 8388608 + 2 * 524288 + 114688 + 4096);     // 1 KB

  prep_kernel<<<1024, 256, 0, stream>>>(W_ih, W_hh, b_ih, b_hh, W_tech, b_tech,
                                        W_phase, b_phase, W_time, b_time,
                                        wih, wcomb, wout, bg, bo);
  proj_kernel<<<BATCH / 64, 256, 0, stream>>>(noise, W_proj, b_proj, x0);

  size_t shmem = (size_t)2 * BUF_ELEMS * sizeof(unsigned short)      // 132 KB h ping-pong
               + (size_t)TB * CSTRIDE * sizeof(float);               // 130 KB cell state
  lstm_gen_kernel<<<BATCH / TB, 256, shmem, stream>>>(x0, wih, wcomb, wout, bg, bo,
                                                      (float*)d_out);
}
